// CoconBlock_27195732918829
// MI455X (gfx1250) — matrix operand order repaired and checked by reading, not compiled, not run on hardware
//
#include <hip/hip_runtime.h>
#include <cstdint>

// ---------------- problem constants ----------------
constexpr int kB = 2, kS = 2048, kC = 512, kD = 768, kH = 12, kDH = 64;
constexpr int kDFF = 3072;
constexpr int kNS = kC + kS;          // 2560 keys
constexpr int kM  = kB * kS;          // 4096 token rows
constexpr float kNEG = 10000.0f;

// ---------------- WMMA fragment types ----------------
typedef __attribute__((ext_vector_type(16))) _Float16 v16h;
typedef __attribute__((ext_vector_type(8)))  _Float16 v8h;
typedef __attribute__((ext_vector_type(8)))  float    v8f;

union FragH { v16h v; v8h h[2]; };

// A-operand (16x32 f16, M x K): lane<16 holds row=lane, K {0..7,16..23};
// lane>=16 same row, K {8..15,24..31}.  tile points at element [0][k0].
__device__ __forceinline__ v16h frag_a(const _Float16* tile, size_t ld, int lane) {
  const _Float16* p = tile + (size_t)(lane & 15) * ld + ((lane >> 4) << 3);
  FragH f;
  f.h[0] = *(const v8h*)p;
  f.h[1] = *(const v8h*)(p + 16);
  return f.v;
}

// B-operand (32x16 f16, K x N) loaded from B^T (N x K, k-contiguous):
// lane<16 holds col n=lane&15, K {0..7,16..23}; lane>=16 K {8..15,24..31}.
__device__ __forceinline__ v16h frag_b(const _Float16* tileT, size_t ld, int lane) {
  const _Float16* p = tileT + (size_t)(lane & 15) * ld + ((lane >> 4) << 3);
  FragH f;
  f.h[0] = *(const v8h*)p;
  f.h[1] = *(const v8h*)(p + 16);
  return f.v;
}

__device__ __forceinline__ v8f wmma32(v16h a, v16h b, v8f c) {
  return __builtin_amdgcn_wmma_f32_16x16x32_f16(false, a, false, b, (short)0, c,
                                                false, false);
}

// ---- CDNA5 async Global->LDS copies (ASYNCcnt-tracked, no VGPR data) ----
// LDS destination = low 32 bits of the generic address (flat-aperture rule).
__device__ __forceinline__ void async_g2l_b128(void* lds, const void* gptr) {
  uint32_t lo = (uint32_t)(uintptr_t)lds;
  uint64_t ga = (uint64_t)(uintptr_t)gptr;
  asm volatile("global_load_async_to_lds_b128 %0, %1, off"
               :: "v"(lo), "v"(ga) : "memory");
}
__device__ __forceinline__ void async_g2l_b64(void* lds, const void* gptr) {
  uint32_t lo = (uint32_t)(uintptr_t)lds;
  uint64_t ga = (uint64_t)(uintptr_t)gptr;
  asm volatile("global_load_async_to_lds_b64 %0, %1, off"
               :: "v"(lo), "v"(ga) : "memory");
}
__device__ __forceinline__ void wait_async0() {
  asm volatile("s_wait_asynccnt 0x0" ::: "memory");
}

// ---------------- small utility kernels ----------------
__global__ __launch_bounds__(256) void cvt_f32_f16(const float* __restrict__ in,
                                                   _Float16* __restrict__ out, int n) {
  int i = blockIdx.x * 256 + threadIdx.x;
  if (i < n) out[i] = (_Float16)in[i];
}

// W[K][N] fp32 -> Wt[N][K] f16 (k-contiguous for WMMA B operand)
__global__ __launch_bounds__(256) void transpose_cvt(const float* __restrict__ W,
                                                     _Float16* __restrict__ Wt,
                                                     int K, int N) {
  size_t i = (size_t)blockIdx.x * 256 + threadIdx.x;
  if (i < (size_t)K * N) {
    size_t k = i / N, n = i % N;
    Wt[n * (size_t)K + k] = (_Float16)W[i];
  }
}

// LayerNorm: one 256-thread block per row of length kD, output f16
__global__ __launch_bounds__(256) void ln_kernel(const float* __restrict__ x,
                                                 const float* __restrict__ g,
                                                 const float* __restrict__ bta,
                                                 _Float16* __restrict__ y) {
  __shared__ float rbuf[256];
  const int tid = threadIdx.x;
  const size_t row = blockIdx.x;
  const float* xr = x + row * kD;
  float s = 0.f;
  for (int i = tid; i < kD; i += 256) s += xr[i];
  rbuf[tid] = s; __syncthreads();
  for (int o = 128; o; o >>= 1) { if (tid < o) rbuf[tid] += rbuf[tid + o]; __syncthreads(); }
  const float mu = rbuf[0] / (float)kD; __syncthreads();
  float v = 0.f;
  for (int i = tid; i < kD; i += 256) { float d = xr[i] - mu; v += d * d; }
  rbuf[tid] = v; __syncthreads();
  for (int o = 128; o; o >>= 1) { if (tid < o) rbuf[tid] += rbuf[tid + o]; __syncthreads(); }
  const float rstd = rsqrtf(rbuf[0] / (float)kD + 1e-5f);
  for (int i = tid; i < kD; i += 256)
    y[row * kD + i] = (_Float16)((xr[i] - mu) * rstd * g[i] + bta[i]);
}

// ---------------- double-buffered async WMMA GEMM ----------------
// A f16 MxK row-major, Bt f16 NxK row-major.  256 threads = 8 waves.
// Block tile 64(M) x 128(N); wave tile 16 x 64 (4 WMMA accumulators,
// one shared A fragment); K step 32; async Global->LDS double buffering.
template <typename Epi>
__global__ __launch_bounds__(256) void gemm_f16_wmma(const _Float16* __restrict__ A,
                                                     int lda,
                                                     const _Float16* __restrict__ Bt,
                                                     int ldbt,
                                                     int M, int N, int K, Epi epi) {
  __shared__ _Float16 sA[2][64 * 32];    // 2 x 4KB
  __shared__ _Float16 sB[2][128 * 32];   // 2 x 8KB
  const int tid  = threadIdx.x;
  const int wid  = tid >> 5, lane = tid & 31;
  const int wm   = wid & 3,  wn   = wid >> 2;          // wn in {0,1}
  const int bm   = blockIdx.x * 64, bn = blockIdx.y * 128;

  auto stage = [&](int buf, int k0) {
    {  // A tile: 64x32 halves = 256 lanes x 16B
      int idx = tid * 8, r = idx >> 5, c = idx & 31;
      async_g2l_b128(&sA[buf][idx], &A[(size_t)(bm + r) * lda + k0 + c]);
    }
#pragma unroll
    for (int t = 0; t < 2; ++t) {  // B tile: 128x32 halves = 512 x 16B
      int idx = (tid + t * 256) * 8, r = idx >> 5, c = idx & 31;
      async_g2l_b128(&sB[buf][idx], &Bt[(size_t)(bn + r) * ldbt + k0 + c]);
    }
  };

  v8f acc[4] = {{}, {}, {}, {}};
  stage(0, 0);
  wait_async0();
  __syncthreads();
  int buf = 0;
  for (int k0 = 0; k0 < K; k0 += 32) {
    if (k0 + 32 < K) stage(buf ^ 1, k0 + 32);          // prefetch next tile
    v16h a = frag_a(&sA[buf][wm * 16 * 32], 32, lane); // shared across 4 WMMAs
#pragma unroll
    for (int nt = 0; nt < 4; ++nt) {
      v16h b = frag_b(&sB[buf][(wn * 64 + nt * 16) * 32], 32, lane);
      acc[nt] = wmma32(a, b, acc[nt]);
    }
    wait_async0();
    __syncthreads();
    buf ^= 1;
  }
  // C/D layout: VGPR r -> row (r + 8*(lane>=16)), col lane&15
  const int col = lane & 15, rb = (lane >> 4) * 8;
  const int m0 = bm + wm * 16;
#pragma unroll
  for (int nt = 0; nt < 4; ++nt) {
    const int n0 = bn + wn * 64 + nt * 16;
#pragma unroll
    for (int r = 0; r < 8; ++r) {
      int m = m0 + rb + r, n = n0 + col;
      if (m < M && n < N) epi(m, n, acc[nt][r]);
    }
  }
}

// ---------------- epilogue functors ----------------
struct EpiQKV {  // rows = b*S+s over x; cols = 3*D; scatter q / k(+C) / v^T
  const float* bias; _Float16 *q, *k, *vT;
  __device__ void operator()(int m, int n, float a) const {
    float t = a + bias[n];
    int b = m / kS, s = m % kS;
    int seg = n / kD, nn = n - seg * kD;
    int h = nn >> 6, d = nn & 63;
    size_t bh = (size_t)b * kH + h;
    if (seg == 0)      q [(bh * kS + s) * kDH + d]            = (_Float16)t;
    else if (seg == 1) k [(bh * kNS + kC + s) * kDH + d]      = (_Float16)t;
    else               vT[(bh * kDH + d) * kNS + (kC + s)]    = (_Float16)t;
  }
};

struct EpiCTX {  // rows = b*C+c; cols = 2*D; scatter kc / vc^T
  const float* bias; _Float16 *k, *vT;
  __device__ void operator()(int m, int n, float a) const {
    float t = a + bias[n];
    int b = m / kC, c = m % kC;
    int seg = n / kD, nn = n - seg * kD;
    int h = nn >> 6, d = nn & 63;
    size_t bh = (size_t)b * kH + h;
    if (seg == 0) k [(bh * kNS + c) * kDH + d] = (_Float16)t;
    else          vT[(bh * kDH + d) * kNS + c] = (_Float16)t;
  }
};

struct EpiProjRes {  // x1 = x + (attn @ Wproj + bproj), fp32 out
  const float* bias; const float* x; float* x1;
  __device__ void operator()(int m, int n, float a) const {
    size_t i = (size_t)m * kD + n;
    x1[i] = a + bias[n] + x[i];
  }
};

struct EpiSilu {  // m1 = silu(h2 @ W1 + b1), f16 out
  const float* bias; _Float16* m1;
  __device__ void operator()(int m, int n, float a) const {
    float t = a + bias[n];
    m1[(size_t)m * kDFF + n] = (_Float16)(t / (1.0f + expf(-t)));
  }
};

struct EpiOut {  // out[:,1:,:] = x1 + (m1 @ W2 + b2), drop s==0
  const float* bias; const float* x1; float* out;
  __device__ void operator()(int m, int n, float a) const {
    float t = a + bias[n] + x1[(size_t)m * kD + n];
    int b = m / kS, s = m % kS;
    if (s > 0) out[((size_t)b * (kS - 1) + (s - 1)) * kD + n] = t;
  }
};

// ---------------- fused attention kernel ----------------
// grid = (S/16, B*H); 256 threads = 8 waves.
// LDS: scores f32 16x2560 | probs f16 16x2560 | q f16 16x64 | red 256 f32
//      | PV split-K partial 16x64 f32
constexpr size_t ATTN_SC   = (size_t)16 * kNS * sizeof(float);
constexpr size_t ATTN_P    = (size_t)16 * kNS * sizeof(_Float16);
constexpr size_t ATTN_Q    = (size_t)16 * kDH * sizeof(_Float16);
constexpr size_t ATTN_RED  = 256 * sizeof(float);
constexpr size_t ATTN_PART = (size_t)16 * kDH * sizeof(float);
constexpr size_t ATTN_LDS  = ATTN_SC + ATTN_P + ATTN_Q + ATTN_RED + ATTN_PART; // ~247KB

__global__ __launch_bounds__(256) void attn_kernel(const _Float16* __restrict__ q,
                                                   const _Float16* __restrict__ kbuf,
                                                   const _Float16* __restrict__ vT,
                                                   _Float16* __restrict__ aout,
                                                   const float* __restrict__ cab_p,
                                                   const int* __restrict__ selfmask_p) {
  extern __shared__ char smem[];
  float*    sc   = (float*)smem;                               // [16][kNS]
  _Float16* sp   = (_Float16*)(smem + ATTN_SC);                // [16][kNS]
  _Float16* qs   = (_Float16*)(smem + ATTN_SC + ATTN_P);       // [16][64]
  float*    red  = (float*)(smem + ATTN_SC + ATTN_P + ATTN_Q); // [256]
  float*    part = (float*)(smem + ATTN_SC + ATTN_P + ATTN_Q + ATTN_RED); // [16][64]

  const int bh = blockIdx.y;                 // b*H + h
  const int b  = bh / kH, h = bh % kH;
  const int m0 = blockIdx.x * 16;            // query row base
  const int tid = threadIdx.x, wid = tid >> 5, lane = tid & 31;
  const float cab = *cab_p;
  const int selfmask = *selfmask_p;

  // ---- async-stage q tile (contiguous 16x64 block, 256 lanes x 8B) ----
  {
    const _Float16* qg = q + ((size_t)bh * kS + m0) * kDH;
    int i = tid * 4;                         // 1024 halves total
    async_g2l_b64(&qs[i], &qg[i]);
    wait_async0();
  }
  __syncthreads();

  // ---- scores = q @ K^T with masks/bias, into LDS fp32 ----
  const _Float16* kb = kbuf + (size_t)bh * kNS * kDH;
  for (int nt = wid; nt < kNS / 16; nt += 8) {
    const int n0 = nt * 16;
    v8f acc = {};
#pragma unroll
    for (int k0 = 0; k0 < kDH; k0 += 32) {
      v16h a  = frag_a(qs + k0, kDH, lane);
      v16h bb = frag_b(kb + (size_t)n0 * kDH + k0, kDH, lane);
      acc = wmma32(a, bb, acc);
    }
    const int j = n0 + (lane & 15), rb = (lane >> 4) * 8;
#pragma unroll
    for (int r = 0; r < 8; ++r) {
      const int qi = m0 + rb + r;
      float w = acc[r];
      const bool keep = (j <= qi + kC) && (!selfmask || j != qi);
      w = keep ? w : -kNEG;
      if (j < kC) w += cab;
      sc[(size_t)(rb + r) * kNS + j] = w;
    }
  }
  __syncthreads();

  // ---- softmax over kNS per row (16 threads per row) ----
  {
    const int row = tid >> 4, sub = tid & 15;
    float* srow = sc + (size_t)row * kNS;
    float mx = -3.4e38f;
    for (int j = sub; j < kNS; j += 16) mx = fmaxf(mx, srow[j]);
    red[tid] = mx; __syncthreads();
    for (int o = 8; o; o >>= 1) { if (sub < o) red[tid] = fmaxf(red[tid], red[tid + o]); __syncthreads(); }
    mx = red[row * 16]; __syncthreads();
    float sm = 0.f;
    for (int j = sub; j < kNS; j += 16) { float e = expf(srow[j] - mx); srow[j] = e; sm += e; }
    red[tid] = sm; __syncthreads();
    for (int o = 8; o; o >>= 1) { if (sub < o) red[tid] += red[tid + o]; __syncthreads(); }
    const float rs = 1.0f / red[row * 16]; __syncthreads();
    for (int j = sub; j < kNS; j += 16)
      sp[(size_t)row * kNS + j] = (_Float16)(srow[j] * rs);
  }
  __syncthreads();

  // ---- out tile = P @ V, split-K across 8 waves ----
  // wave w: n-tile (w&3) of DH=64, K half (w>>2) of kNS.
  {
    const int n0 = (wid & 3) * 16;
    const int khalf = wid >> 2;
    const int kbeg = khalf * (kNS / 2), kend = kbeg + (kNS / 2);
    const _Float16* vb = vT + (size_t)bh * kDH * kNS;
    v8f acc = {};
    for (int k0 = kbeg; k0 < kend; k0 += 32) {
      v16h a  = frag_a(sp + k0, kNS, lane);
      v16h bb = frag_b(vb + (size_t)n0 * kNS + k0, kNS, lane);
      acc = wmma32(a, bb, acc);
    }
    const int col = lane & 15, rb = (lane >> 4) * 8;
    if (khalf == 1) {
#pragma unroll
      for (int r = 0; r < 8; ++r) part[(rb + r) * kDH + n0 + col] = acc[r];
    }
    __syncthreads();
    if (khalf == 0) {
#pragma unroll
      for (int r = 0; r < 8; ++r) {
        float t = acc[r] + part[(rb + r) * kDH + n0 + col];
        aout[((size_t)b * kS + (m0 + rb + r)) * kD + h * kDH + n0 + col] = (_Float16)t;
      }
    }
  }
}

// ---------------- host-side orchestration ----------------
extern "C" void kernel_launch(void* const* d_in, const int* in_sizes, int n_in,
                              void* d_out, int out_size, void* d_ws, size_t ws_size,
                              hipStream_t stream) {
  const float* x       = (const float*)d_in[0];
  const float* ctx     = (const float*)d_in[1];
  const float* ln1_g   = (const float*)d_in[2];
  const float* ln1_b   = (const float*)d_in[3];
  const float* Wattn   = (const float*)d_in[4];
  const float* battn   = (const float*)d_in[5];
  const float* Wref    = (const float*)d_in[6];
  const float* bref    = (const float*)d_in[7];
  const float* Wproj   = (const float*)d_in[8];
  const float* bproj   = (const float*)d_in[9];
  const float* ln2_g   = (const float*)d_in[10];
  const float* ln2_b   = (const float*)d_in[11];
  const float* W1      = (const float*)d_in[12];
  const float* b1      = (const float*)d_in[13];
  const float* W2      = (const float*)d_in[14];
  const float* b2      = (const float*)d_in[15];
  const float* cab     = (const float*)d_in[16];
  const int*   smaskp  = (const int*)d_in[17];
  float* out = (float*)d_out;

  char* ws = (char*)d_ws;
  size_t off = 0;
  auto carve = [&](size_t bytes) -> void* {
    void* p = ws + off;
    off += (bytes + 255) & ~(size_t)255;
    return p;
  };
  _Float16* h1     = (_Float16*)carve((size_t)kM * kD * 2);
  _Float16* ctx16  = (_Float16*)carve((size_t)kB * kC * kD * 2);
  _Float16* WattnT = (_Float16*)carve((size_t)3 * kD * kD * 2);
  _Float16* WrefT  = (_Float16*)carve((size_t)2 * kD * kD * 2);
  _Float16* WprojT = (_Float16*)carve((size_t)kD * kD * 2);
  _Float16* W1T    = (_Float16*)carve((size_t)kDFF * kD * 2);
  _Float16* W2T    = (_Float16*)carve((size_t)kD * kDFF * 2);
  _Float16* qb     = (_Float16*)carve((size_t)kB * kH * kS * kDH * 2);
  _Float16* kb     = (_Float16*)carve((size_t)kB * kH * kNS * kDH * 2);
  _Float16* vTb    = (_Float16*)carve((size_t)kB * kH * kDH * kNS * 2);
  _Float16* aout   = (_Float16*)carve((size_t)kM * kD * 2);
  float*    x1     = (float*)   carve((size_t)kM * kD * 4);
  _Float16* h2     = (_Float16*)carve((size_t)kM * kD * 2);
  _Float16* m1     = (_Float16*)carve((size_t)kM * kDFF * 2);

  // 1. weight conversion (fp32 KxN -> f16 NxK)
  auto tgrid = [](size_t n) { return dim3((unsigned)((n + 255) / 256)); };
  transpose_cvt<<<tgrid((size_t)kD * 3 * kD), 256, 0, stream>>>(Wattn, WattnT, kD, 3 * kD);
  transpose_cvt<<<tgrid((size_t)kD * 2 * kD), 256, 0, stream>>>(Wref,  WrefT,  kD, 2 * kD);
  transpose_cvt<<<tgrid((size_t)kD * kD),     256, 0, stream>>>(Wproj, WprojT, kD, kD);
  transpose_cvt<<<tgrid((size_t)kD * kDFF),   256, 0, stream>>>(W1,    W1T,    kD, kDFF);
  transpose_cvt<<<tgrid((size_t)kDFF * kD),   256, 0, stream>>>(W2,    W2T,    kDFF, kD);
  cvt_f32_f16<<<tgrid((size_t)kB * kC * kD), 256, 0, stream>>>(ctx, ctx16, kB * kC * kD);

  // 2. LN1
  ln_kernel<<<dim3(kM), 256, 0, stream>>>(x, ln1_g, ln1_b, h1);

  // 3. fused QKV GEMM + context KV GEMM
  gemm_f16_wmma<EpiQKV><<<dim3(kM / 64, (3 * kD) / 128), 256, 0, stream>>>(
      h1, kD, WattnT, kD, kM, 3 * kD, kD, EpiQKV{battn, qb, kb, vTb});
  gemm_f16_wmma<EpiCTX><<<dim3((kB * kC) / 64, (2 * kD) / 128), 256, 0, stream>>>(
      ctx16, kD, WrefT, kD, kB * kC, 2 * kD, kD, EpiCTX{bref, kb, vTb});

  // 4. fused attention (scores + softmax + split-K PV in LDS)
  attn_kernel<<<dim3(kS / 16, kB * kH), 256, ATTN_LDS, stream>>>(
      qb, kb, vTb, aout, cab, smaskp);

  // 5. output projection + residual
  gemm_f16_wmma<EpiProjRes><<<dim3(kM / 64, kD / 128), 256, 0, stream>>>(
      aout, kD, WprojT, kD, kM, kD, kD, EpiProjRes{bproj, x, x1});

  // 6. LN2 + FFN + residual + token drop
  ln_kernel<<<dim3(kM), 256, 0, stream>>>(x1, ln2_g, ln2_b, h2);
  gemm_f16_wmma<EpiSilu><<<dim3(kM / 64, kDFF / 128), 256, 0, stream>>>(
      h2, kD, W1T, kD, kM, kDFF, kD, EpiSilu{b1, m1});
  gemm_f16_wmma<EpiOut><<<dim3(kM / 64, kD / 128), 256, 0, stream>>>(
      m1, kDFF, W2T, kDFF, kM, kD, kDFF, EpiOut{b2, x1, out});
}
